// BRNNFeatureFusion_43404939493545
// MI455X (gfx1250) — matrix operand order repaired and checked by reading, not compiled, not run on hardware
//
#include <hip/hip_runtime.h>
#include <hip/hip_bf16.h>

// ---------------------------------------------------------------------------
// BRNN feature fusion for MI455X (gfx1250, wave32, WMMA).
// f16 operands / f32 accum via v_wmma_f32_16x16x32_f16.
// Weights pre-transposed to N-major so every WMMA fragment is b128 loads.
// ---------------------------------------------------------------------------

typedef __attribute__((ext_vector_type(16))) _Float16 v16h;
typedef __attribute__((ext_vector_type(8)))  _Float16 v8h;
typedef __attribute__((ext_vector_type(8)))  float    v8f;

#define WMMA16(A, Bv, C) \
    __builtin_amdgcn_wmma_f32_16x16x32_f16(false, (A), false, (Bv), (short)0, (C), false, false)

static constexpr int BB = 1024;     // batch
static constexpr int TT = 128;      // time steps
static constexpr int FF = 256;      // features
static constexpr int UU = 256;      // rnn units
static constexpr int G3 = 3 * UU;   // 768
static constexpr int BT = BB * TT;  // 131072

// ---- fragment loaders (per WMMA VGPR layout, CDNA5 ISA 7.12.2) --
// A (16x32 f16): lane holds K = {base..base+7} and {base+16..base+23}
__device__ __forceinline__ v16h ldA16(const _Float16* __restrict__ p) {
    v8h lo = *(const v8h*)(p);
    v8h hi = *(const v8h*)(p + 16);
    return __builtin_shufflevector(lo, hi, 0, 1, 2, 3, 4, 5, 6, 7, 8, 9, 10, 11, 12, 13, 14, 15);
}
// B (32x16 f16, N-major source): lane holds K = {base..base+7} and {base+16..base+23}
__device__ __forceinline__ v16h ldB16(const _Float16* __restrict__ p) {
    v8h lo = *(const v8h*)(p);
    v8h hi = *(const v8h*)(p + 16);
    return __builtin_shufflevector(lo, hi, 0, 1, 2, 3, 4, 5, 6, 7, 8, 9, 10, 11, 12, 13, 14, 15);
}
// A from f32 source: two aligned float4 pairs + cvt
__device__ __forceinline__ v16h ldA32(const float* __restrict__ p) {
    float4 a0 = *(const float4*)(p);
    float4 a1 = *(const float4*)(p + 4);
    float4 b0 = *(const float4*)(p + 16);
    float4 b1 = *(const float4*)(p + 20);
    v16h v;
    v[0] = (_Float16)a0.x;  v[1] = (_Float16)a0.y;  v[2] = (_Float16)a0.z;  v[3] = (_Float16)a0.w;
    v[4] = (_Float16)a1.x;  v[5] = (_Float16)a1.y;  v[6] = (_Float16)a1.z;  v[7] = (_Float16)a1.w;
    v[8] = (_Float16)b0.x;  v[9] = (_Float16)b0.y;  v[10] = (_Float16)b0.z; v[11] = (_Float16)b0.w;
    v[12] = (_Float16)b1.x; v[13] = (_Float16)b1.y; v[14] = (_Float16)b1.z; v[15] = (_Float16)b1.w;
    return v;
}

// ---------------------------------------------------------------------------
// Weight prep: f32 [K,N] -> f16 [N,K] (transpose-convert)
// ---------------------------------------------------------------------------
__global__ void k_cvtT(const float* __restrict__ src, _Float16* __restrict__ dst, int K, int N) {
    int i = blockIdx.x * 256 + threadIdx.x;
    if (i < K * N) {
        int k = i / N, n = i - k * N;
        dst[(size_t)n * K + k] = (_Float16)src[i];
    }
}
// wf [192,11] -> padded transposed f16 [16,192] (rows >= 11 zero)
__global__ void k_pad_wfT(const float* __restrict__ wf, _Float16* __restrict__ wfT) {
    int i = blockIdx.x * 256 + threadIdx.x;  // 16*192
    if (i < 16 * 192) {
        int n = i / 192, k = i - n * 192;
        wfT[i] = (n < 11) ? (_Float16)wf[k * 11 + n] : (_Float16)0.0f;
    }
}

// ---------------------------------------------------------------------------
// xp = einsum('btf,fu->tbu', x[, ::-1], wx) + b  -> f16 [T, B, 768]
// rows = t*B + b. 4 waves/block, each wave: 16 rows x 32 cols (2 N-tiles).
// ---------------------------------------------------------------------------
__global__ void k_gemm_xp(const float* __restrict__ x, const _Float16* __restrict__ wxT,
                          const float* __restrict__ bx, _Float16* __restrict__ xp, int reverse) {
    const int lane = threadIdx.x & 31, wave = threadIdx.x >> 5;
    const int ln15 = lane & 15, hi = lane >> 4;
    const int mBase = (blockIdx.x * 4 + wave) * 16;
    const int nBase = blockIdx.y * 32;

    const int row = mBase + ln15;                  // t*B + b
    const int b = row & (BB - 1);
    const int t = row >> 10;                       // B == 1024
    const int tx = reverse ? (TT - 1 - t) : t;
    const float* __restrict__ xrow = x + ((size_t)b * TT + tx) * FF + hi * 8;
    const _Float16* __restrict__ bp0 = wxT + (size_t)(nBase + ln15) * FF + hi * 8;
    const _Float16* __restrict__ bp1 = bp0 + (size_t)16 * FF;

    v8f acc0 = {}, acc1 = {};
#pragma unroll
    for (int kk = 0; kk < FF; kk += 32) {
        __builtin_prefetch(xrow + kk + 96, 0, 1);  // global_prefetch_b8
        v16h a = ldA32(xrow + kk);
        acc0 = WMMA16(a, ldB16(bp0 + kk), acc0);
        acc1 = WMMA16(a, ldB16(bp1 + kk), acc1);
    }
    const int c0 = nBase + ln15, c1 = c0 + 16;
    const float bias0 = bx[c0], bias1 = bx[c1];
#pragma unroll
    for (int r = 0; r < 8; ++r) {
        int mr = mBase + r + (hi << 3);
        xp[(size_t)mr * G3 + c0] = (_Float16)(acc0[r] + bias0);
        xp[(size_t)mr * G3 + c1] = (_Float16)(acc1[r] + bias1);
    }
}

// ---------------------------------------------------------------------------
// One GRU step, both directions (blockIdx.z). Each wave: 16 batch x 16 units,
// three WMMA chains (z/r/h blocks) sharing the A fragment, fused gate math.
// ---------------------------------------------------------------------------
__global__ void k_gru_step(const _Float16* __restrict__ xpF, const _Float16* __restrict__ xpB,
                           const _Float16* __restrict__ whTF, const _Float16* __restrict__ whTB,
                           const float* __restrict__ bhF, const float* __restrict__ bhB,
                           _Float16* __restrict__ ysF, _Float16* __restrict__ ysB, int s) {
    const int dir = blockIdx.z;
    const _Float16* __restrict__ xp_t = (dir == 0 ? xpF : xpB) + (size_t)s * BB * G3;
    const _Float16* __restrict__ whT = dir == 0 ? whTF : whTB;
    const float* __restrict__ bh = dir == 0 ? bhF : bhB;
    _Float16* __restrict__ ys = dir == 0 ? ysF : ysB;
    const int t_prev = (dir == 0) ? (s - 1) : ((s == 0) ? -1 : (TT - s));
    const int t_cur  = (dir == 0) ? s : (TT - 1 - s);

    const int lane = threadIdx.x & 31, wave = threadIdx.x >> 5;
    const int ln15 = lane & 15, hi = lane >> 4;
    const int mBase = (blockIdx.x * 4 + wave) * 16;   // batch tile
    const int uBase = blockIdx.y * 16;                // unit tile

    const int u = uBase + ln15;
    const _Float16* __restrict__ pz = whT + (size_t)u * UU + hi * 8;
    const _Float16* __restrict__ pr = whT + (size_t)(UU + u) * UU + hi * 8;
    const _Float16* __restrict__ ph = whT + (size_t)(2 * UU + u) * UU + hi * 8;

    v8f cz = {}, cr = {}, ch = {};
    if (t_prev >= 0) {
        const _Float16* __restrict__ hrow =
            ys + ((size_t)(mBase + ln15) * TT + t_prev) * UU + hi * 8;
#pragma unroll
        for (int kk = 0; kk < UU; kk += 32) {
            __builtin_prefetch(hrow + kk + 96, 0, 1);
            v16h a = ldA16(hrow + kk);
            cz = WMMA16(a, ldB16(pz + kk), cz);
            cr = WMMA16(a, ldB16(pr + kk), cr);
            ch = WMMA16(a, ldB16(ph + kk), ch);
        }
    }

    const float bzs = bh[u], brs = bh[UU + u], bhs = bh[2 * UU + u];
#pragma unroll
    for (int r = 0; r < 8; ++r) {
        int brow = mBase + r + (hi << 3);
        size_t xb = (size_t)brow * G3;
        float xz = (float)xp_t[xb + u];
        float xr = (float)xp_t[xb + UU + u];
        float xh = (float)xp_t[xb + 2 * UU + u];
        float z  = 1.0f / (1.0f + __expf(-(xz + cz[r] + bzs)));
        float rg = 1.0f / (1.0f + __expf(-(xr + cr[r] + brs)));
        float hp = (t_prev >= 0) ? (float)ys[((size_t)brow * TT + t_prev) * UU + u] : 0.0f;
        float hh = fmaxf(0.0f, xh + rg * (ch[r] + bhs));
        float hn = z * hp + (1.0f - z) * hh;
        ys[((size_t)brow * TT + t_cur) * UU + u] = (_Float16)hn;
    }
}

// ---------------------------------------------------------------------------
// fnn1 = relu(x @ w1 + b1)  -> f16 [BT, 64]
// ---------------------------------------------------------------------------
__global__ void k_fnn1(const float* __restrict__ x, const _Float16* __restrict__ w1T,
                       const float* __restrict__ b1, _Float16* __restrict__ fnn1) {
    const int lane = threadIdx.x & 31, wave = threadIdx.x >> 5;
    const int ln15 = lane & 15, hi = lane >> 4;
    const int mBase = (blockIdx.x * 4 + wave) * 16;
    const int nBase = blockIdx.y * 16;
    const float* __restrict__ xrow = x + (size_t)(mBase + ln15) * FF + hi * 8;
    const _Float16* __restrict__ bp = w1T + (size_t)(nBase + ln15) * FF + hi * 8;

    v8f acc = {};
#pragma unroll
    for (int kk = 0; kk < FF; kk += 32) {
        __builtin_prefetch(xrow + kk + 96, 0, 1);
        acc = WMMA16(ldA32(xrow + kk), ldB16(bp + kk), acc);
    }
    const int col = nBase + ln15;
    const float bias = b1[col];
#pragma unroll
    for (int r = 0; r < 8; ++r) {
        int mr = mBase + r + (hi << 3);
        fnn1[(size_t)mr * 64 + col] = (_Float16)fmaxf(0.0f, acc[r] + bias);
    }
}

// ---------------------------------------------------------------------------
// out128 = concat( relu([hf|hb] @ w0 + b0), relu(fnn1 @ w2 + b2) ) f32
// ---------------------------------------------------------------------------
__global__ void k_fuse128(const _Float16* __restrict__ ysf, const _Float16* __restrict__ ysb,
                          const _Float16* __restrict__ fnn1,
                          const _Float16* __restrict__ w0T, const _Float16* __restrict__ w2T,
                          const float* __restrict__ b0, const float* __restrict__ b2,
                          float* __restrict__ out128) {
    const int lane = threadIdx.x & 31, wave = threadIdx.x >> 5;
    const int ln15 = lane & 15, hi = lane >> 4;
    const int mBase = (blockIdx.x * 4 + wave) * 16;
    const int nTile = blockIdx.y;
    const size_t rowA = (size_t)(mBase + ln15);

    v8f acc = {};
    if (nTile < 4) {
        const _Float16* __restrict__ fr = ysf + rowA * UU + hi * 8;
        const _Float16* __restrict__ br = ysb + rowA * UU + hi * 8;
        const _Float16* __restrict__ bp = w0T + (size_t)(nTile * 16 + ln15) * 512 + hi * 8;
#pragma unroll
        for (int kk = 0; kk < 2 * UU; kk += 32) {
            const _Float16* src = (kk < UU) ? (fr + kk) : (br + (kk - UU));
            acc = WMMA16(ldA16(src), ldB16(bp + kk), acc);
        }
    } else {
        const _Float16* __restrict__ ar = fnn1 + rowA * 64 + hi * 8;
        const _Float16* __restrict__ bp = w2T + (size_t)((nTile - 4) * 16 + ln15) * 64 + hi * 8;
#pragma unroll
        for (int kk = 0; kk < 64; kk += 32)
            acc = WMMA16(ldA16(ar + kk), ldB16(bp + kk), acc);
    }
    const int c = nTile * 16 + ln15;
    const float bias = (nTile < 4) ? b0[c] : b2[c - 64];
#pragma unroll
    for (int r = 0; r < 8; ++r) {
        int mr = mBase + r + (hi << 3);
        out128[(size_t)mr * 128 + c] = fmaxf(0.0f, acc[r] + bias);
    }
}

// ---------------------------------------------------------------------------
// dense3: t3 = relu(out1n @ w3 + b3) f32 [BT, 64]
// ---------------------------------------------------------------------------
__global__ void k_dense3(const float* __restrict__ out1n, const _Float16* __restrict__ w3T,
                         const float* __restrict__ b3, float* __restrict__ t3) {
    const int lane = threadIdx.x & 31, wave = threadIdx.x >> 5;
    const int ln15 = lane & 15, hi = lane >> 4;
    const int mBase = (blockIdx.x * 4 + wave) * 16;
    const int nBase = blockIdx.y * 16;
    const float* __restrict__ arow = out1n + (size_t)(mBase + ln15) * 128 + hi * 8;
    const _Float16* __restrict__ bp = w3T + (size_t)(nBase + ln15) * 128 + hi * 8;

    v8f acc = {};
#pragma unroll
    for (int kk = 0; kk < 128; kk += 32) {
        __builtin_prefetch(arow + kk + 96, 0, 1);
        acc = WMMA16(ldA32(arow + kk), ldB16(bp + kk), acc);
    }
    const int col = nBase + ln15;
    const float bias = b3[col];
#pragma unroll
    for (int r = 0; r < 8; ++r) {
        int mr = mBase + r + (hi << 3);
        t3[(size_t)mr * 64 + col] = fmaxf(0.0f, acc[r] + bias);
    }
}

// ---------------------------------------------------------------------------
// BatchNorm (training): stats over batch axis per (t, c); LDS tree reduce.
// ---------------------------------------------------------------------------
__global__ void k_bn(const float* __restrict__ y, float* __restrict__ out,
                     const float* __restrict__ gamma, const float* __restrict__ beta, int C) {
    __shared__ float s1[256];
    __shared__ float s2[256];
    const int tc = blockIdx.x;
    const int c = tc % C, t = tc / C;
    float v[4], sum = 0.0f, sq = 0.0f;
#pragma unroll
    for (int i = 0; i < 4; ++i) {
        int b = threadIdx.x + i * 256;
        v[i] = y[((size_t)b * TT + t) * C + c];
        sum += v[i];
        sq += v[i] * v[i];
    }
    s1[threadIdx.x] = sum;
    s2[threadIdx.x] = sq;
    __syncthreads();
    for (int off = 128; off > 0; off >>= 1) {
        if (threadIdx.x < off) {
            s1[threadIdx.x] += s1[threadIdx.x + off];
            s2[threadIdx.x] += s2[threadIdx.x + off];
        }
        __syncthreads();
    }
    const float mean = s1[0] * (1.0f / 1024.0f);
    const float var  = s2[0] * (1.0f / 1024.0f) - mean * mean;
    const float scl  = gamma[c] * rsqrtf(var + 1e-3f);
    const float sh   = beta[c] - mean * scl;
#pragma unroll
    for (int i = 0; i < 4; ++i) {
        int b = threadIdx.x + i * 256;
        out[((size_t)b * TT + t) * C + c] = v[i] * scl + sh;
    }
}

// ---------------------------------------------------------------------------
// final: softmax( [out1n | out2n] @ wf + bf ) -> f32 [BT, 11]
// wfT padded [16,192]; softmax via width-16 shfl reductions.
// ---------------------------------------------------------------------------
__global__ void k_final(const float* __restrict__ out1n, const float* __restrict__ out2n,
                        const _Float16* __restrict__ wfT, const float* __restrict__ bf,
                        float* __restrict__ dout) {
    const int lane = threadIdx.x & 31, wave = threadIdx.x >> 5;
    const int ln15 = lane & 15, hi = lane >> 4;
    const int mBase = (blockIdx.x * 4 + wave) * 16;
    const size_t rowA = (size_t)(mBase + ln15);
    const float* __restrict__ a1 = out1n + rowA * 128 + hi * 8;
    const float* __restrict__ a2 = out2n + rowA * 64 + hi * 8;
    const _Float16* __restrict__ bp = wfT + (size_t)ln15 * 192 + hi * 8;

    v8f acc = {};
#pragma unroll
    for (int kk = 0; kk < 192; kk += 32) {
        v16h a = (kk < 128) ? ldA32(a1 + kk) : ldA32(a2 + (kk - 128));
        acc = WMMA16(a, ldB16(bp + kk), acc);
    }
    const int n = ln15;
#pragma unroll
    for (int r = 0; r < 8; ++r) {
        int m = mBase + r + (hi << 3);
        float val = (n < 11) ? (acc[r] + bf[n]) : -3.0e38f;
        float vmax = val;
#pragma unroll
        for (int off = 1; off < 16; off <<= 1)
            vmax = fmaxf(vmax, __shfl_xor(vmax, off, 16));
        float e = (n < 11) ? __expf(val - vmax) : 0.0f;
        float s = e;
#pragma unroll
        for (int off = 1; off < 16; off <<= 1)
            s += __shfl_xor(s, off, 16);
        if (n < 11) dout[(size_t)m * 11 + n] = e / s;
    }
}

// ---------------------------------------------------------------------------
// Host orchestration
// ---------------------------------------------------------------------------
extern "C" void kernel_launch(void* const* d_in, const int* in_sizes, int n_in,
                              void* d_out, int out_size, void* d_ws, size_t ws_size,
                              hipStream_t stream) {
    (void)in_sizes; (void)n_in; (void)out_size; (void)ws_size;

    const float* x    = (const float*)d_in[0];
    const float* wx_f = (const float*)d_in[1];
    const float* wh_f = (const float*)d_in[2];
    const float* b_f  = (const float*)d_in[3];   // [2,768]
    const float* wx_b = (const float*)d_in[4];
    const float* wh_b = (const float*)d_in[5];
    const float* b_b  = (const float*)d_in[6];
    const float* w0   = (const float*)d_in[7];
    const float* b0   = (const float*)d_in[8];
    const float* w1   = (const float*)d_in[9];
    const float* b1   = (const float*)d_in[10];
    const float* w2   = (const float*)d_in[11];
    const float* b2   = (const float*)d_in[12];
    const float* w3   = (const float*)d_in[13];
    const float* b3   = (const float*)d_in[14];
    const float* wf   = (const float*)d_in[15];
    const float* bfv  = (const float*)d_in[16];
    const float* g1   = (const float*)d_in[17];
    const float* be1  = (const float*)d_in[18];
    const float* g2   = (const float*)d_in[19];
    const float* be2  = (const float*)d_in[20];
    float* dout = (float*)d_out;

    size_t off = 0;
    auto alloc = [&](size_t bytes) -> void* {
        off = (off + 255) & ~(size_t)255;
        void* p = (char*)d_ws + off;
        off += bytes;
        return p;
    };
    _Float16* wxTf = (_Float16*)alloc((size_t)FF * G3 * 2);
    _Float16* whTf = (_Float16*)alloc((size_t)UU * G3 * 2);
    _Float16* wxTb = (_Float16*)alloc((size_t)FF * G3 * 2);
    _Float16* whTb = (_Float16*)alloc((size_t)UU * G3 * 2);
    _Float16* w0T  = (_Float16*)alloc((size_t)512 * 64 * 2);
    _Float16* w1T  = (_Float16*)alloc((size_t)256 * 64 * 2);
    _Float16* w2T  = (_Float16*)alloc((size_t)64 * 64 * 2);
    _Float16* w3T  = (_Float16*)alloc((size_t)128 * 64 * 2);
    _Float16* wfT  = (_Float16*)alloc((size_t)16 * 192 * 2);
    _Float16* xpF  = (_Float16*)alloc((size_t)BT * G3 * 2);
    _Float16* xpB  = (_Float16*)alloc((size_t)BT * G3 * 2);
    _Float16* ysF  = (_Float16*)alloc((size_t)BT * UU * 2);
    _Float16* ysB  = (_Float16*)alloc((size_t)BT * UU * 2);
    _Float16* fnn1 = (_Float16*)alloc((size_t)BT * 64 * 2);
    float*    o128 = (float*)alloc((size_t)BT * 128 * 4);
    float*    o1n  = (float*)alloc((size_t)BT * 128 * 4);
    float*    t3   = (float*)alloc((size_t)BT * 64 * 4);
    float*    o2n  = (float*)alloc((size_t)BT * 64 * 4);

    auto cvtT = [&](const float* s, _Float16* d, int K, int N) {
        k_cvtT<<<(K * N + 255) / 256, 256, 0, stream>>>(s, d, K, N);
    };
    cvtT(wx_f, wxTf, FF, G3);
    cvtT(wh_f, whTf, UU, G3);
    cvtT(wx_b, wxTb, FF, G3);
    cvtT(wh_b, whTb, UU, G3);
    cvtT(w0, w0T, 512, 64);
    cvtT(w1, w1T, 256, 64);
    cvtT(w2, w2T, 64, 64);
    cvtT(w3, w3T, 128, 64);
    k_pad_wfT<<<(16 * 192 + 255) / 256, 256, 0, stream>>>(wf, wfT);

    // input projections (both directions), rows = t*B + b, 2 N-tiles per wave
    dim3 gXP(BT / 64, G3 / 32);
    k_gemm_xp<<<gXP, 128, 0, stream>>>(x, wxTf, b_f, xpF, 0);
    k_gemm_xp<<<gXP, 128, 0, stream>>>(x, wxTb, b_b, xpB, 1);

    // GRU recurrence: 128 steps, both directions per launch
    dim3 gStep(BB / 64, UU / 16, 2);
    for (int s = 0; s < TT; ++s) {
        k_gru_step<<<gStep, 128, 0, stream>>>(xpF, xpB, whTf, whTb,
                                              b_f + G3, b_b + G3, ysF, ysB, s);
    }

    // dense branch + fusion
    k_fnn1<<<dim3(BT / 64, 4), 128, 0, stream>>>(x, w1T, b1, fnn1);
    k_fuse128<<<dim3(BT / 64, 8), 128, 0, stream>>>(ysF, ysB, fnn1, w0T, w2T, b0, b2, o128);

    // BN1 -> dense3 -> BN2
    k_bn<<<TT * 128, 256, 0, stream>>>(o128, o1n, g1, be1, 128);
    k_dense3<<<dim3(BT / 64, 4), 128, 0, stream>>>(o1n, w3T, b3, t3);
    k_bn<<<TT * 64, 256, 0, stream>>>(t3, o2n, g2, be2, 64);

    // final classifier + softmax
    k_final<<<BT / 64, 128, 0, stream>>>(o1n, o2n, wfT, bfv, dout);
}
